// AttnBlock_9225589751870
// MI455X (gfx1250) — matrix operand order repaired and checked by reading, not compiled, not run on hardware
//
#include <hip/hip_runtime.h>
#include <hip/hip_bf16.h>

#define DEVINL __device__ __forceinline__

typedef __attribute__((ext_vector_type(16))) _Float16 v16h;
typedef __attribute__((ext_vector_type(8)))  _Float16 v8h;
typedef __attribute__((ext_vector_type(8)))  float    v8f;
typedef __attribute__((ext_vector_type(4)))  int      v4i;

constexpr int Cc = 512;    // channels
constexpr int Nn = 1024;   // h*w
constexpr int Bb = 16;     // batch

// ---------------- LDS panel geometry -----------------------------------
// Block tile 128(M) x 128(N), K-step 32.  Panels: 128 rows x 32 halves,
// padded to 40 halves (80 B) pitch -> every fragment sub-load stays 16 B
// aligned while rows stagger across banks.  Double buffered.
constexpr int PAD   = 40;
constexpr int PANEL = 128 * PAD;          // halves per buffer (10 KB)

#ifndef __has_builtin
#define __has_builtin(x) 0
#endif
#if __has_builtin(__builtin_amdgcn_global_load_async_to_lds_b128)
#define ASYNC_LDS 1
#else
#define ASYNC_LDS 0
#endif

// copy 16 bytes global -> LDS (async on CDNA5 when available)
DEVINL void copy16(const _Float16* __restrict__ g, _Float16* l) {
#if ASYNC_LDS
  __builtin_amdgcn_global_load_async_to_lds_b128(
      (__attribute__((address_space(1))) v4i*)(g),
      (__attribute__((address_space(3))) v4i*)(l), 0, 0);
#else
  *(v8h*)l = *(const v8h*)g;
#endif
}

DEVINL void wait_async_copies() {
#if ASYNC_LDS
  asm volatile("s_wait_asynccnt 0" ::: "memory");
#endif
}

// ---------------- WMMA fragments from LDS -------------------------------
// A 16x32 (MxK): lane l -> row (l&15); halves K = hf*8+{0..7}, 16+hf*8+{0..7}
DEVINL v16h frag_a_at(const _Float16* p) {
  v8h lo = *(const v8h*)p;
  v8h hi = *(const v8h*)(p + 16);
  return __builtin_shufflevector(lo, hi, 0,1,2,3,4,5,6,7,8,9,10,11,12,13,14,15);
}
// B 32x16 (KxN) from BT rows: lane l -> col (l&15); halves K = hf*8+{0..7}, 16+hf*8+{0..7}
DEVINL v16h frag_b_at(const _Float16* p) {
  v8h lo = *(const v8h*)p;
  v8h hi = *(const v8h*)(p + 16);
  return __builtin_shufflevector(lo, hi, 0,1,2,3,4,5,6,7,8,9,10,11,12,13,14,15);
}

DEVINL v8f wmma_f16(v16h a, v16h b, v8f c) {
  return __builtin_amdgcn_wmma_f32_16x16x32_f16(false, a, false, b, (short)0, c, false, false);
}

// ---------------- block GEMM: 128x128 tile, 8 waves of 32x64 ------------
// A (MxK row-major, lda), BT (NxK row-major, ldb), f32 accumulate.
// acc[mi*4+ni] = 16x16 subtile at (wm+16mi, wn+16ni) within the block tile.
template<int KTOT>
DEVINL void block_gemm(const _Float16* __restrict__ A, int lda,
                       const _Float16* __restrict__ BT, int ldb,
                       int mbase, int nbase, int tid,
                       _Float16* ldsA, _Float16* ldsB, v8f acc[8]) {
  int lane = tid & 31, wave = tid >> 5;
  int wm = (wave >> 1) * 32;                 // 0,32,64,96
  int wn = (wave & 1) * 64;                  // 0,64
  constexpr int KS = KTOT / 32;

  // ---- fixed per-thread staging slots: 2 x 16B chunks per panel --------
  const int srow = tid >> 2;                 // 0..63
  const int scol = (tid & 3) * 8;            // 0,8,16,24 (halves)
  const _Float16* gA0 = A  + (size_t)(mbase + srow) * lda + scol;
  const _Float16* gA1 = gA0 + (size_t)64 * lda;
  const _Float16* gB0 = BT + (size_t)(nbase + srow) * ldb + scol;
  const _Float16* gB1 = gB0 + (size_t)64 * ldb;
  _Float16* lA = ldsA + srow * PAD + scol;
  _Float16* lB = ldsB + srow * PAD + scol;

  // ---- fixed per-wave fragment base pointers ---------------------------
  const int fr = lane & 15, fh = lane >> 4;
  const _Float16* aBase = ldsA + (wm + fr) * PAD + fh * 8;
  const _Float16* bBase = ldsB + (wn + fr) * PAD + fh * 8;

  // prologue: stage K-panel 0 into buffer 0
  copy16(gA0, lA);
  copy16(gA1, lA + 64 * PAD);
  copy16(gB0, lB);
  copy16(gB1, lB + 64 * PAD);
  wait_async_copies();
  __syncthreads();

  #pragma unroll 2
  for (int ks = 0; ks < KS; ++ks) {
    const int cur = ks & 1;
    if (ks + 1 < KS) {                       // prefetch next K-panel (other buffer)
      const int off = (ks + 1) * 32;         // halves
      const int lo  = (cur ^ 1) * PANEL;
      copy16(gA0 + off, lA + lo);
      copy16(gA1 + off, lA + lo + 64 * PAD);
      copy16(gB0 + off, lB + lo);
      copy16(gB1 + off, lB + lo + 64 * PAD);
    }
    const _Float16* Ap = aBase + cur * PANEL;
    const _Float16* Bp = bBase + cur * PANEL;
    v16h a0 = frag_a_at(Ap);
    v16h a1 = frag_a_at(Ap + 16 * PAD);
    v16h b0 = frag_b_at(Bp);
    v16h b1 = frag_b_at(Bp + 16 * PAD);
    v16h b2 = frag_b_at(Bp + 32 * PAD);
    v16h b3 = frag_b_at(Bp + 48 * PAD);
    acc[0] = wmma_f16(a0, b0, acc[0]);
    acc[1] = wmma_f16(a0, b1, acc[1]);
    acc[2] = wmma_f16(a0, b2, acc[2]);
    acc[3] = wmma_f16(a0, b3, acc[3]);
    acc[4] = wmma_f16(a1, b0, acc[4]);
    acc[5] = wmma_f16(a1, b1, acc[5]);
    acc[6] = wmma_f16(a1, b2, acc[6]);
    acc[7] = wmma_f16(a1, b3, acc[7]);
    wait_async_copies();
    __syncthreads();
  }
}

// Store one 16x16 f32 D-tile as f16 into a transposed (N-major, M contiguous)
// destination: one b128 store per lane.  D: VGPR i, lane l -> M=i+8*hf, N=l&15.
DEVINL void store_f16_T(_Float16* __restrict__ DT, int ldt, int m0, int n0, int lane, v8f acc) {
  int r = lane & 15, hf = lane >> 4;
  v8h h;
  #pragma unroll
  for (int i = 0; i < 8; ++i) h[i] = (_Float16)acc[i];
  *(v8h*)(DT + (size_t)(n0 + r) * ldt + m0 + hf * 8) = h;
}

// ---------------------------------------------------------------------------
// Kernel 0: convert the four 512x512 f32 weights to f16 (row-major o,c)
// ---------------------------------------------------------------------------
__global__ __launch_bounds__(256) void cvt_w_kernel(
    const float* __restrict__ wq, const float* __restrict__ wk,
    const float* __restrict__ wv, const float* __restrict__ wo,
    _Float16* __restrict__ w16) {
  int i = blockIdx.x * 256 + threadIdx.x;   // 0 .. C*C-1
  w16[0 * Cc * Cc + i] = (_Float16)wq[i];
  w16[1 * Cc * Cc + i] = (_Float16)wk[i];
  w16[2 * Cc * Cc + i] = (_Float16)wv[i];
  w16[3 * Cc * Cc + i] = (_Float16)wo[i];
}

// ---------------------------------------------------------------------------
// Kernel 1: GroupNorm (32 groups, 16 ch/group). One block per (b, g).
// Writes hnT[b][n][c] f16 (c contiguous) = K-contiguous WMMA operand layout.
// ---------------------------------------------------------------------------
__global__ __launch_bounds__(256) void groupnorm_kernel(
    const float* __restrict__ x, const float* __restrict__ gscale,
    const float* __restrict__ gbias, _Float16* __restrict__ hnT) {
  int b = blockIdx.x >> 5;
  int g = blockIdx.x & 31;
  const float* xg = x + ((size_t)b * Cc + g * 16) * Nn;   // 16 channels x Nn
  int tid = threadIdx.x;

  float s = 0.f, s2 = 0.f;
  for (int idx = tid; idx < 16 * Nn; idx += 256) {        // coalesced in n
    float v = xg[idx];
    s += v; s2 += v * v;
  }
  __shared__ float rs[256], rs2[256];
  rs[tid] = s; rs2[tid] = s2;
  __syncthreads();
  for (int off = 128; off > 0; off >>= 1) {
    if (tid < off) { rs[tid] += rs[tid + off]; rs2[tid] += rs2[tid + off]; }
    __syncthreads();
  }
  float mean = rs[0] * (1.0f / 16384.0f);
  float var  = rs2[0] * (1.0f / 16384.0f) - mean * mean;
  float rstd = rsqrtf(var + 1e-6f);

  for (int idx = tid; idx < 16 * Nn; idx += 256) {        // coalesced f16 writes
    int cc = idx & 15;
    int n  = idx >> 4;
    int c  = g * 16 + cc;
    float v = (xg[(size_t)cc * Nn + n] - mean) * rstd;
    hnT[((size_t)b * Nn + n) * Cc + c] = (_Float16)(v * gscale[c] + gbias[c]);
  }
}

// ---------------------------------------------------------------------------
// Kernel 2: Q/K/V projections.  Out[o,n] = sum_c W[o,c]*hn[c,n] + bias[o]
// Q,K stored transposed (n,c); V stored (c,n).
// ---------------------------------------------------------------------------
__global__ __launch_bounds__(256) void proj_qkv_kernel(
    const _Float16* __restrict__ w16, const float* __restrict__ bq,
    const float* __restrict__ bk, const float* __restrict__ bv,
    const _Float16* __restrict__ hnT,
    _Float16* __restrict__ qT, _Float16* __restrict__ kT, _Float16* __restrict__ vm) {
  __shared__ _Float16 sA[2 * PANEL];
  __shared__ _Float16 sB[2 * PANEL];
  int tid = threadIdx.x, lane = tid & 31, wave = tid >> 5;
  int mat = blockIdx.y, b = blockIdx.z;
  int mbase = (blockIdx.x >> 3) * 128;       // 4 m-blocks over C
  int nbase = (blockIdx.x & 7) * 128;        // 8 n-blocks over N

  const _Float16* A  = w16 + (size_t)mat * Cc * Cc;
  const _Float16* BT = hnT + (size_t)b * Nn * Cc;
  v8f acc[8] = {};
  block_gemm<Cc>(A, Cc, BT, Cc, mbase, nbase, tid, sA, sB, acc);

  const float* bias = (mat == 0) ? bq : ((mat == 1) ? bk : bv);
  int r = lane & 15, hf = lane >> 4;
  int wm = (wave >> 1) * 32, wn = (wave & 1) * 64;

  if (mat < 2) {
    _Float16* DT = (mat == 0 ? qT : kT) + (size_t)b * Nn * Cc;   // (n, o)
    #pragma unroll
    for (int mi = 0; mi < 2; ++mi)
      #pragma unroll
      for (int ni = 0; ni < 4; ++ni) {
        int m0 = mbase + wm + mi * 16, n0 = nbase + wn + ni * 16;
        v8f a = acc[mi * 4 + ni];
        #pragma unroll
        for (int i = 0; i < 8; ++i) a[i] += bias[m0 + hf * 8 + i];
        store_f16_T(DT, Cc, m0, n0, lane, a);
      }
  } else {
    _Float16* V = vm + (size_t)b * Cc * Nn;                      // (c, n)
    #pragma unroll
    for (int mi = 0; mi < 2; ++mi)
      #pragma unroll
      for (int ni = 0; ni < 4; ++ni) {
        int m0 = mbase + wm + mi * 16, n0 = nbase + wn + ni * 16;
        #pragma unroll
        for (int i = 0; i < 8; ++i) {
          int o = m0 + hf * 8 + i;
          V[(size_t)o * Nn + n0 + r] = (_Float16)(acc[mi * 4 + ni][i] + bias[o]);
        }
      }
  }
}

// ---------------------------------------------------------------------------
// Kernel 3: scores[b,i,j] = sum_c q[c,i]*k[c,j]  (scale applied in softmax)
// ---------------------------------------------------------------------------
__global__ __launch_bounds__(256) void scores_kernel(
    const _Float16* __restrict__ qT, const _Float16* __restrict__ kT,
    float* __restrict__ sc) {
  __shared__ _Float16 sA[2 * PANEL];
  __shared__ _Float16 sB[2 * PANEL];
  int tid = threadIdx.x, lane = tid & 31, wave = tid >> 5;
  int b = blockIdx.z;
  int mbase = (blockIdx.x >> 3) * 128;       // 8 x 8 blocks over N x N
  int nbase = (blockIdx.x & 7) * 128;
  v8f acc[8] = {};
  block_gemm<Cc>(qT + (size_t)b * Nn * Cc, Cc, kT + (size_t)b * Nn * Cc, Cc,
                 mbase, nbase, tid, sA, sB, acc);
  float* S = sc + (size_t)b * Nn * Nn;
  int r = lane & 15, hf = lane >> 4;
  int wm = (wave >> 1) * 32, wn = (wave & 1) * 64;
  #pragma unroll
  for (int mi = 0; mi < 2; ++mi)
    #pragma unroll
    for (int ni = 0; ni < 4; ++ni) {
      int i0 = mbase + wm + mi * 16, j0 = nbase + wn + ni * 16;
      #pragma unroll
      for (int i = 0; i < 8; ++i)
        S[(size_t)(i0 + hf * 8 + i) * Nn + j0 + r] = acc[mi * 4 + ni][i];
    }
}

// ---------------------------------------------------------------------------
// Kernel 4: softmax over j (row length 1024), scale = C^-0.5. f32 -> f16.
// ---------------------------------------------------------------------------
__global__ __launch_bounds__(256) void softmax_kernel(
    const float* __restrict__ sc, _Float16* __restrict__ attn) {
  int row = blockIdx.x;                       // b*Nn + i
  const float* s = sc + (size_t)row * Nn;
  _Float16* a = attn + (size_t)row * Nn;
  int tid = threadIdx.x;
  const float scale = 0.04419417382415922f;   // 512^-0.5

  float vals[4];
  float m = -3.4e38f;
  #pragma unroll
  for (int t = 0; t < 4; ++t) {
    vals[t] = s[tid + t * 256] * scale;
    m = fmaxf(m, vals[t]);
  }
  __shared__ float red[256];
  red[tid] = m; __syncthreads();
  for (int off = 128; off > 0; off >>= 1) {
    if (tid < off) red[tid] = fmaxf(red[tid], red[tid + off]);
    __syncthreads();
  }
  m = red[0];
  __syncthreads();

  float sum = 0.f;
  #pragma unroll
  for (int t = 0; t < 4; ++t) { vals[t] = __expf(vals[t] - m); sum += vals[t]; }
  red[tid] = sum; __syncthreads();
  for (int off = 128; off > 0; off >>= 1) {
    if (tid < off) red[tid] += red[tid + off];
    __syncthreads();
  }
  float inv = 1.0f / red[0];
  #pragma unroll
  for (int t = 0; t < 4; ++t) a[tid + t * 256] = (_Float16)(vals[t] * inv);
}

// ---------------------------------------------------------------------------
// Kernel 5: O[c,i] = sum_j V[c,j]*attn[i,j].  Store OT (i,c).
// ---------------------------------------------------------------------------
__global__ __launch_bounds__(256) void av_kernel(
    const _Float16* __restrict__ vm, const _Float16* __restrict__ attn,
    _Float16* __restrict__ oT) {
  __shared__ _Float16 sA[2 * PANEL];
  __shared__ _Float16 sB[2 * PANEL];
  int tid = threadIdx.x, lane = tid & 31, wave = tid >> 5;
  int b = blockIdx.z;
  int mbase = (blockIdx.x >> 3) * 128;       // 4 x 8 blocks over C x N
  int nbase = (blockIdx.x & 7) * 128;
  v8f acc[8] = {};
  block_gemm<Nn>(vm + (size_t)b * Cc * Nn, Nn, attn + (size_t)b * Nn * Nn, Nn,
                 mbase, nbase, tid, sA, sB, acc);
  _Float16* DT = oT + (size_t)b * Nn * Cc;   // (i, c)
  int wm = (wave >> 1) * 32, wn = (wave & 1) * 64;
  #pragma unroll
  for (int mi = 0; mi < 2; ++mi)
    #pragma unroll
    for (int ni = 0; ni < 4; ++ni)
      store_f16_T(DT, Cc, mbase + wm + mi * 16, nbase + wn + ni * 16, lane,
                  acc[mi * 4 + ni]);
}

// ---------------------------------------------------------------------------
// Kernel 6: out = x + wo @ O + bo. f32 output with residual.
// ---------------------------------------------------------------------------
__global__ __launch_bounds__(256) void final_kernel(
    const _Float16* __restrict__ w16, const float* __restrict__ bo,
    const _Float16* __restrict__ oT, const float* __restrict__ x,
    float* __restrict__ out) {
  __shared__ _Float16 sA[2 * PANEL];
  __shared__ _Float16 sB[2 * PANEL];
  int tid = threadIdx.x, lane = tid & 31, wave = tid >> 5;
  int b = blockIdx.z;
  int mbase = (blockIdx.x >> 3) * 128;       // 4 x 8 blocks over C x N
  int nbase = (blockIdx.x & 7) * 128;
  v8f acc[8] = {};
  block_gemm<Cc>(w16 + (size_t)3 * Cc * Cc, Cc, oT + (size_t)b * Nn * Cc, Cc,
                 mbase, nbase, tid, sA, sB, acc);
  int r = lane & 15, hf = lane >> 4;
  int wm = (wave >> 1) * 32, wn = (wave & 1) * 64;
  #pragma unroll
  for (int mi = 0; mi < 2; ++mi)
    #pragma unroll
    for (int ni = 0; ni < 4; ++ni) {
      int m0 = mbase + wm + mi * 16, n0 = nbase + wn + ni * 16;
      #pragma unroll
      for (int i = 0; i < 8; ++i) {
        int o = m0 + hf * 8 + i;
        size_t idx = ((size_t)b * Cc + o) * Nn + n0 + r;
        out[idx] = x[idx] + acc[mi * 4 + ni][i] + bo[o];
      }
    }
}

// ---------------------------------------------------------------------------
extern "C" void kernel_launch(void* const* d_in, const int* in_sizes, int n_in,
                              void* d_out, int out_size, void* d_ws, size_t ws_size,
                              hipStream_t stream) {
  const float* x      = (const float*)d_in[0];
  const float* nscale = (const float*)d_in[1];
  const float* nbias  = (const float*)d_in[2];
  const float* wq = (const float*)d_in[3];  const float* bq = (const float*)d_in[4];
  const float* wk = (const float*)d_in[5];  const float* bk = (const float*)d_in[6];
  const float* wv = (const float*)d_in[7];  const float* bv = (const float*)d_in[8];
  const float* wo = (const float*)d_in[9];  const float* bo = (const float*)d_in[10];
  float* out = (float*)d_out;

  // workspace carve-up
  _Float16* w16  = (_Float16*)d_ws;                       //  4*C*C        (2 MB)
  _Float16* hnT  = w16 + (size_t)4 * Cc * Cc;             //  B*N*C        (16 MB)
  _Float16* qT   = hnT + (size_t)Bb * Nn * Cc;
  _Float16* kT   = qT  + (size_t)Bb * Nn * Cc;
  _Float16* vm   = kT  + (size_t)Bb * Nn * Cc;
  _Float16* oT   = vm  + (size_t)Bb * Nn * Cc;
  _Float16* attn = oT  + (size_t)Bb * Nn * Cc;            //  B*N*N        (32 MB)
  float*    sc   = (float*)(attn + (size_t)Bb * Nn * Nn); //  B*N*N f32    (64 MB)

  cvt_w_kernel<<<dim3(Cc * Cc / 256), dim3(256), 0, stream>>>(wq, wk, wv, wo, w16);
  groupnorm_kernel<<<dim3(Bb * 32), dim3(256), 0, stream>>>(x, nscale, nbias, hnT);
  proj_qkv_kernel<<<dim3(32, 3, Bb), dim3(256), 0, stream>>>(w16, bq, bk, bv, hnT, qT, kT, vm);
  scores_kernel<<<dim3(64, 1, Bb), dim3(256), 0, stream>>>(qT, kT, sc);
  softmax_kernel<<<dim3(Bb * Nn), dim3(256), 0, stream>>>(sc, attn);
  av_kernel<<<dim3(32, 1, Bb), dim3(256), 0, stream>>>(vm, attn, oT);
  final_kernel<<<dim3(32, 1, Bb), dim3(256), 0, stream>>>(w16, bo, oT, x, out);
}
